// SelfAttention_50002009260470
// MI455X (gfx1250) — matrix operand order repaired and checked by reading, not compiled, not run on hardware
//
#include <hip/hip_runtime.h>
#include <hip/hip_bf16.h>
#include <stdint.h>

// ---------------------------------------------------------------------------
// Fused self-attention for MI455X (gfx1250).
//   B=2, S=4096, E=768, H=12, D=64.  ~132 GFLOP.  bf16 WMMA + fp32 acc.
//   (1) TDM (tensor_load_to_lds, 6-arg clang-23 builtin) double-buffered K/V
//   tile staging shared by all 8 waves of a block (8x cut in L2 traffic,
//   exercises TENSORcnt), (2) 16x64 GEMM tiles reusing A across 4 WMMAs.
// ---------------------------------------------------------------------------

typedef __attribute__((ext_vector_type(16))) __bf16 v16bf;
typedef __attribute__((ext_vector_type(8)))  __bf16 v8bf;
typedef __attribute__((ext_vector_type(8)))  float  v8f;
typedef __attribute__((ext_vector_type(4)))  unsigned int v4u32;
typedef __attribute__((ext_vector_type(8)))  int v8i32;
typedef __attribute__((ext_vector_type(4)))  int v4i32;

union FragAB { v16bf v; v8bf h[2]; };

#define E_DIM  768
#define S_LEN  4096
#define NH     12
#define HD     64
#define BATCH  2
#define MROWS  (BATCH * S_LEN)   // 8192

// ---------------------------------------------------------------------------
// fp32 -> bf16 elementwise convert
// ---------------------------------------------------------------------------
__global__ void cvt_f32_bf16(const float* __restrict__ in,
                             __hip_bfloat16* __restrict__ out, int n) {
  int i = blockIdx.x * 256 + threadIdx.x;
  if (i < n) out[i] = __float2bfloat16(in[i]);
}

// ---------------------------------------------------------------------------
// TDM: async 2D bf16 tile load Global -> LDS (clang-23 6-arg builtin form:
// (v4u g0, v8i g1, v4i g2, v4i g3, v8i extra, i32 cpol)).
// D# bitfields per CDNA5 ISA:
//  g0: [1:0]=count=1 | lds_addr[63:32] | global_addr[120:64] | type=2[127:126]
//  g1: data_size=1(2B)[17:16] | tensor_dim0[79:48] | tensor_dim1[111:80]
//      | tile_dim0[127:112] | tile_dim1[143:128] | tensor_dim0_stride[207:160]
//  g2/g3/extra: zero (2D tensor, tile_dim2=0)
// ---------------------------------------------------------------------------
__device__ __forceinline__ void tdm_load_2d_bf16(uint32_t lds_off, const void* gptr,
                                                 uint32_t tensor_d0, uint32_t tensor_d1,
                                                 uint32_t tile_d0, uint32_t tile_d1,
                                                 uint64_t stride0_elems) {
  const uint64_t ga = (uint64_t)(uintptr_t)gptr;
  v4u32 g0;
  g0.x = 1u;                                               // count=1 (valid)
  g0.y = lds_off;                                          // LDS byte address
  g0.z = (uint32_t)ga;                                     // global_addr[31:0]
  g0.w = ((uint32_t)(ga >> 32) & 0x01FFFFFFu) | (2u << 30);// [56:32] | type=2
  v8i32 g1;
  g1[0] = (int)(1u << 16);                                 // data_size=2B
  g1[1] = (int)((tensor_d0 & 0xFFFFu) << 16);              // dim0[15:0]
  g1[2] = (int)(((tensor_d0 >> 16) & 0xFFFFu) | ((tensor_d1 & 0xFFFFu) << 16));
  g1[3] = (int)(((tensor_d1 >> 16) & 0xFFFFu) | ((tile_d0 & 0xFFFFu) << 16));
  g1[4] = (int)(tile_d1 & 0xFFFFu);                        // tile_dim1, tile_dim2=0
  g1[5] = (int)(uint32_t)(stride0_elems & 0xFFFFFFFFull);  // dim0_stride[31:0]
  g1[6] = (int)(uint32_t)((stride0_elems >> 32) & 0xFFFFull);
  g1[7] = 0;
  const v4i32 gz4 = {0, 0, 0, 0};
  const v8i32 gz8 = {0, 0, 0, 0, 0, 0, 0, 0};
  __builtin_amdgcn_tensor_load_to_lds(g0, g1, gz4, gz4, gz8, 0);
}

// ---------------------------------------------------------------------------
// Y = X @ W^T + b.  16x64 output tile per wave: one A fragment reused by 4
// WMMAs against 4 B fragments (each a single contiguous 32B load since the
// NT contraction runs along rows of both X and W).
//   out_mode 0: out[b][h][s][d]   (Q, K)
//   out_mode 1: out[b][h][d][s]   (V transposed for contiguous P@V B-frags)
// ---------------------------------------------------------------------------
__global__ void qkv_gemm_wmma(const __bf16* __restrict__ X,
                              const __bf16* __restrict__ W,
                              const float* __restrict__ bias,
                              __hip_bfloat16* __restrict__ out,
                              int out_mode) {
  const int lane = threadIdx.x & 31;
  const int wave = threadIdx.x >> 5;
  const int NG   = E_DIM / 64;                      // 12 column groups
  const int tile = blockIdx.x * 8 + wave;           // 6144 total (768 blocks)
  const int tm   = tile / NG;
  const int tg   = tile - tm * NG;
  if (tm >= MROWS / 16) return;
  const int m0 = tm * 16, n0 = tg * 64;

  const int l16 = lane & 15;
  const int hi  = lane >> 4;

  // A (16x32): row=m0+l16; lanes<16 hold K=0..15 contiguous, lanes>=16 K=16..31
  const __bf16* aBase = X + (size_t)(m0 + l16) * E_DIM + (hi << 4);
  // B (32x16): col=n+l16; lanes<16 hold K=0..15 contiguous, lanes>=16 K=16..31
  const __bf16* bBase0 = W + (size_t)(n0 +  0 + l16) * E_DIM + (hi << 4);
  const __bf16* bBase1 = W + (size_t)(n0 + 16 + l16) * E_DIM + (hi << 4);
  const __bf16* bBase2 = W + (size_t)(n0 + 32 + l16) * E_DIM + (hi << 4);
  const __bf16* bBase3 = W + (size_t)(n0 + 48 + l16) * E_DIM + (hi << 4);

  v8f acc0 = {}, acc1 = {}, acc2 = {}, acc3 = {};
  for (int k0 = 0; k0 < E_DIM; k0 += 32) {
    FragAB a;
    a.h[0] = *(const v8bf*)(aBase + k0);
    a.h[1] = *(const v8bf*)(aBase + k0 + 8);
    v16bf b0 = *(const v16bf*)(bBase0 + k0);
    v16bf b1 = *(const v16bf*)(bBase1 + k0);
    v16bf b2 = *(const v16bf*)(bBase2 + k0);
    v16bf b3 = *(const v16bf*)(bBase3 + k0);
    acc0 = __builtin_amdgcn_wmma_f32_16x16x32_bf16(false, a.v, false, b0, (short)0, acc0, false, false);
    acc1 = __builtin_amdgcn_wmma_f32_16x16x32_bf16(false, a.v, false, b1, (short)0, acc1, false, false);
    acc2 = __builtin_amdgcn_wmma_f32_16x16x32_bf16(false, a.v, false, b2, (short)0, acc2, false, false);
    acc3 = __builtin_amdgcn_wmma_f32_16x16x32_bf16(false, a.v, false, b3, (short)0, acc3, false, false);
  }

  const int h = n0 >> 6;                // whole 64-wide group is one head
  v8f* accs[4] = { &acc0, &acc1, &acc2, &acc3 };
  #pragma unroll
  for (int t = 0; t < 4; ++t) {
    const int   d    = t * 16 + l16;    // C layout: N = lane%16
    const float bval = bias[n0 + d];
    #pragma unroll
    for (int r = 0; r < 8; ++r) {       // C layout: M = r (+8 upper half)
      const int m = m0 + r + (hi << 3);
      const int b = m >> 12;            // m = b*S + s
      const int s = m & (S_LEN - 1);
      size_t idx;
      if (out_mode == 0)
        idx = (((size_t)(b * NH + h)) * S_LEN + s) * HD + d;
      else
        idx = (((size_t)(b * NH + h)) * HD + d) * S_LEN + s;
      out[idx] = __float2bfloat16((*accs[t])[r] + bval);
    }
  }
}

// ---------------------------------------------------------------------------
// Flash attention (no 1/sqrt(D), matching reference).
// Block = 256 threads (8 waves) owning 128 query rows of one (b,h).
// K/V tiles for 32 keys are TDM-loaded into double-buffered LDS by wave 0
// (tensor_load_to_lds + s_wait_tensorcnt + barrier); all 8 waves consume the
// same tiles via ds_load_b128 fragment reads -> 8x less L2 traffic.
// ---------------------------------------------------------------------------
__global__ void flash_attn_wmma(const __bf16* __restrict__ Q,
                                const __bf16* __restrict__ K,
                                const __bf16* __restrict__ Vt,
                                float* __restrict__ out) {
  const int lane = threadIdx.x & 31;
  const int wave = threadIdx.x >> 5;
  const int bh   = blockIdx.x >> 5;        // 32 row-blocks per (b,h)
  const int sblk = blockIdx.x & 31;
  const int b    = bh / NH;
  const int h    = bh - b * NH;
  const int q0   = sblk * 128 + wave * 16;

  const int l16 = lane & 15;
  const int hi  = lane >> 4;

  const __bf16* Qb = Q  + (size_t)bh * S_LEN * HD;   // [S,D]
  const __bf16* Kb = K  + (size_t)bh * S_LEN * HD;   // [S,D]
  const __bf16* Vb = Vt + (size_t)bh * HD * S_LEN;   // [D,S]

  __shared__ __align__(16) __hip_bfloat16 ktile[2][32][64];  // 8 KB
  __shared__ __align__(16) __hip_bfloat16 vtile[2][64][32];  // 8 KB
  __shared__ __align__(16) __hip_bfloat16 plds[8][16][40];   // 10 KB (P relay)

  // Q fragments for d=[0,32) and d=[32,64): loaded once, reused 128 times.
  const __bf16* qrow = Qb + (size_t)(q0 + l16) * HD;
  FragAB qf0, qf1;
  qf0.h[0] = *(const v8bf*)(qrow + (hi << 4));
  qf0.h[1] = *(const v8bf*)(qrow + (hi << 4) + 8);
  qf1.h[0] = *(const v8bf*)(qrow + 32 + (hi << 4));
  qf1.h[1] = *(const v8bf*)(qrow + 32 + (hi << 4) + 8);

  float mrow[8], lrow[8];
  v8f o0 = {}, o1 = {}, o2 = {}, o3 = {};
  #pragma unroll
  for (int r = 0; r < 8; ++r) { mrow[r] = -3.0e38f; lrow[r] = 0.0f; }

  auto issue_chunk = [&](int j0, int par) {
    // K tile: 32 keys x 64 d, rows contiguous (stride 64) -> LDS [32][64]
    tdm_load_2d_bf16((uint32_t)(uintptr_t)&ktile[par][0][0],
                     Kb + (size_t)j0 * HD,
                     /*tensor_d0=*/HD, /*tensor_d1=*/(uint32_t)(S_LEN - j0),
                     /*tile_d0=*/HD, /*tile_d1=*/32, /*stride0=*/HD);
    // V tile: 64 d-rows x 32 j-cols from Vt[D,S] (stride 4096) -> LDS [64][32]
    tdm_load_2d_bf16((uint32_t)(uintptr_t)&vtile[par][0][0],
                     Vb + j0,
                     /*tensor_d0=*/(uint32_t)(S_LEN - j0), /*tensor_d1=*/HD,
                     /*tile_d0=*/32, /*tile_d1=*/HD, /*stride0=*/S_LEN);
  };

  if (wave == 0) issue_chunk(0, 0);

  const int NCHUNK = S_LEN / 32;   // 128
  for (int c = 0; c < NCHUNK; ++c) {
    const int par = c & 1;
    if (wave == 0) {
      if (c + 1 < NCHUNK) {
        issue_chunk((c + 1) * 32, par ^ 1);   // prefetch next into other buf
        __builtin_amdgcn_s_wait_tensorcnt(2); // chunk c's 2 loads are done
      } else {
        __builtin_amdgcn_s_wait_tensorcnt(0);
      }
    }
    __syncthreads();   // LDS tiles for chunk c visible to all waves

    // --- scores: S[16 x 32] = Q @ K_tile^T (K frags from LDS) ---
    v16bf ka0 = *(const v16bf*)&ktile[par][     l16][(hi << 4)];
    v16bf ka1 = *(const v16bf*)&ktile[par][     l16][(hi << 4) + 32];
    v16bf kb0 = *(const v16bf*)&ktile[par][16 + l16][(hi << 4)];
    v16bf kb1 = *(const v16bf*)&ktile[par][16 + l16][(hi << 4) + 32];

    v8f sa = {}, sb = {};
    sa = __builtin_amdgcn_wmma_f32_16x16x32_bf16(false, qf0.v, false, ka0, (short)0, sa, false, false);
    sa = __builtin_amdgcn_wmma_f32_16x16x32_bf16(false, qf1.v, false, ka1, (short)0, sa, false, false);
    sb = __builtin_amdgcn_wmma_f32_16x16x32_bf16(false, qf0.v, false, kb0, (short)0, sb, false, false);
    sb = __builtin_amdgcn_wmma_f32_16x16x32_bf16(false, qf1.v, false, kb1, (short)0, sb, false, false);

    // --- online softmax: rows live across the 16 lanes of each wave half ---
    float pa[8], pb[8], alpha[8];
    #pragma unroll
    for (int r = 0; r < 8; ++r) {
      float t = fmaxf(sa[r], sb[r]);
      t = fmaxf(t, __shfl_xor(t, 1, 32));
      t = fmaxf(t, __shfl_xor(t, 2, 32));
      t = fmaxf(t, __shfl_xor(t, 4, 32));
      t = fmaxf(t, __shfl_xor(t, 8, 32));
      const float nm = fmaxf(mrow[r], t);
      alpha[r] = __expf(mrow[r] - nm);
      pa[r]    = __expf(sa[r] - nm);
      pb[r]    = __expf(sb[r] - nm);
      float rs = pa[r] + pb[r];
      rs += __shfl_xor(rs, 1, 32);
      rs += __shfl_xor(rs, 2, 32);
      rs += __shfl_xor(rs, 4, 32);
      rs += __shfl_xor(rs, 8, 32);
      lrow[r] = lrow[r] * alpha[r] + rs;
      mrow[r] = nm;
    }
    #pragma unroll
    for (int r = 0; r < 8; ++r) {
      o0[r] *= alpha[r]; o1[r] *= alpha[r];
      o2[r] *= alpha[r]; o3[r] *= alpha[r];
    }

    // --- relay P from C-layout to A-fragment layout through LDS ---
    #pragma unroll
    for (int r = 0; r < 8; ++r) {
      const int row = r + (hi << 3);
      plds[wave][row][l16]      = __float2bfloat16(pa[r]);
      plds[wave][row][16 + l16] = __float2bfloat16(pb[r]);
    }
    asm volatile("s_wait_dscnt 0" ::: "memory");  // cross-lane LDS RAW

    FragAB pf;  // A 16x32: row = l16; K contiguous at col hi*16 .. hi*16+15
    pf.h[0] = *(const v8bf*)(&plds[wave][l16][(hi << 4)]);
    pf.h[1] = *(const v8bf*)(&plds[wave][l16][(hi << 4) + 8]);

    // --- ctx: O[16 x 64] += P @ V_tile (V frags from LDS, j contiguous) ---
    v16bf v0 = *(const v16bf*)&vtile[par][ 0 + l16][(hi << 4)];
    v16bf v1 = *(const v16bf*)&vtile[par][16 + l16][(hi << 4)];
    v16bf v2 = *(const v16bf*)&vtile[par][32 + l16][(hi << 4)];
    v16bf v3 = *(const v16bf*)&vtile[par][48 + l16][(hi << 4)];

    o0 = __builtin_amdgcn_wmma_f32_16x16x32_bf16(false, pf.v, false, v0, (short)0, o0, false, false);
    o1 = __builtin_amdgcn_wmma_f32_16x16x32_bf16(false, pf.v, false, v1, (short)0, o1, false, false);
    o2 = __builtin_amdgcn_wmma_f32_16x16x32_bf16(false, pf.v, false, v2, (short)0, o2, false, false);
    o3 = __builtin_amdgcn_wmma_f32_16x16x32_bf16(false, pf.v, false, v3, (short)0, o3, false, false);

    __syncthreads();   // everyone done reading buf[par] before it is refilled
  }

  // --- epilogue: out[b, s, h*64 + d] = O / l ---
  #pragma unroll
  for (int r = 0; r < 8; ++r) {
    const int    srow = q0 + r + (hi << 3);
    const size_t base = ((size_t)b * S_LEN + srow) * (NH * HD) + h * HD + l16;
    const float  inv  = 1.0f / lrow[r];
    out[base +  0] = o0[r] * inv;
    out[base + 16] = o1[r] * inv;
    out[base + 32] = o2[r] * inv;
    out[base + 48] = o3[r] * inv;
  }
}

// ---------------------------------------------------------------------------
extern "C" void kernel_launch(void* const* d_in, const int* in_sizes, int n_in,
                              void* d_out, int out_size, void* d_ws, size_t ws_size,
                              hipStream_t stream) {
  const float* hs = (const float*)d_in[0];
  const float* Wq = (const float*)d_in[1];
  const float* bq = (const float*)d_in[2];
  const float* Wk = (const float*)d_in[3];
  const float* bk = (const float*)d_in[4];
  const float* Wv = (const float*)d_in[5];
  const float* bv = (const float*)d_in[6];

  const size_t nX = (size_t)MROWS * E_DIM;     // 6,291,456
  const size_t nW = (size_t)E_DIM * E_DIM;     //   589,824
  const size_t nQ = nX;

  size_t off = 0;
  auto carve = [&](size_t elems) {
    void* p = (char*)d_ws + off;
    off += (elems * sizeof(__hip_bfloat16) + 255) & ~(size_t)255;
    return (__hip_bfloat16*)p;
  };
  __hip_bfloat16* Xbf  = carve(nX);
  __hip_bfloat16* Wqb  = carve(nW);
  __hip_bfloat16* Wkb  = carve(nW);
  __hip_bfloat16* Wvb  = carve(nW);
  __hip_bfloat16* Qbf  = carve(nQ);
  __hip_bfloat16* Kbf  = carve(nQ);
  __hip_bfloat16* Vtbf = carve(nQ);

  cvt_f32_bf16<<<(int)((nX + 255) / 256), 256, 0, stream>>>(hs, Xbf, (int)nX);
  cvt_f32_bf16<<<(int)((nW + 255) / 256), 256, 0, stream>>>(Wq, Wqb, (int)nW);
  cvt_f32_bf16<<<(int)((nW + 255) / 256), 256, 0, stream>>>(Wk, Wkb, (int)nW);
  cvt_f32_bf16<<<(int)((nW + 255) / 256), 256, 0, stream>>>(Wv, Wvb, (int)nW);

  // 16x64 tiles: 512 * 12 = 6144 waves -> 768 blocks of 8 waves.
  qkv_gemm_wmma<<<768, 256, 0, stream>>>((const __bf16*)Xbf, (const __bf16*)Wqb, bq, Qbf, 0);
  qkv_gemm_wmma<<<768, 256, 0, stream>>>((const __bf16*)Xbf, (const __bf16*)Wkb, bk, Kbf, 0);
  qkv_gemm_wmma<<<768, 256, 0, stream>>>((const __bf16*)Xbf, (const __bf16*)Wvb, bv, Vtbf, 1);

  // flash attention: (B*H)=24 x (S/128)=32 -> 768 blocks.
  flash_attn_wmma<<<768, 256, 0, stream>>>((const __bf16*)Qbf, (const __bf16*)Kbf,
                                           (const __bf16*)Vtbf, (float*)d_out);
}
